// TransGNN_54941221650865
// MI455X (gfx1250) — matrix operand order repaired and checked by reading, not compiled, not run on hardware
//
#include <hip/hip_runtime.h>
#include <hip/hip_bf16.h>
#include <stdint.h>

// ---------------------------------------------------------------------------
// TransGNN forward for MI455X (gfx1250, wave32, v_wmma_f32_16x16x32_bf16).
// - Weights pre-packed into per-lane-contiguous B-fragment layout (2x b128).
// - A-fragments: per-lane contiguous per ISA 16-bit A layout (2x b128).
// - GEMM waves own a 32-row M tile (2 WMMAs per B fragment load).
// - No separate fp32->bf16 passes: every producer also emits bf16.
// - Attention fused per node (one wave, K=12 padded to M=16 WMMA tile).
// ---------------------------------------------------------------------------

typedef __attribute__((ext_vector_type(16))) __bf16 v16bf;
typedef __attribute__((ext_vector_type(8)))  float  v8f;
typedef __attribute__((ext_vector_type(8)))  int    v8i;
typedef __attribute__((ext_vector_type(4)))  int    v4i;

#define KSAMP 12

__device__ __forceinline__ uint16_t f2bf(float f) {
  uint32_t u = __float_as_uint(f);
  uint32_t r = u + (((u >> 16) & 1u) + 0x7fffu);   // round-to-nearest-even
  return (uint16_t)(r >> 16);
}
__device__ __forceinline__ uint32_t pack2bf(float a, float b) {
  return (uint32_t)f2bf(a) | ((uint32_t)f2bf(b) << 16);
}

__device__ __forceinline__ v8f wmma_bf16(v8i a_raw, v8i b_raw, v8f acc) {
  return __builtin_amdgcn_wmma_f32_16x16x32_bf16(
      false, __builtin_bit_cast(v16bf, a_raw),
      false, __builtin_bit_cast(v16bf, b_raw),
      (short)0, acc, false, false);
}

__device__ __forceinline__ v8i cat44(v4i lo, v4i hi) {
  v8i r;
#pragma unroll
  for (int i = 0; i < 4; ++i) { r[i] = lo[i]; r[i + 4] = hi[i]; }
  return r;
}

// A fragment (16x32 bf16, row-major source). Per ISA layout a lane's 8 dwords
// are two contiguous 16B runs: K = khalf+[0,8) and K = 16+khalf+[0,8).
__device__ __forceinline__ v8i load_afrag(const uint16_t* __restrict__ Abf,
                                          long rowbase, int kt, int lane) {
  const uint16_t* p = Abf + rowbase + (long)kt * 32 + ((lane >> 4) << 3);
  v4i lo = *(const v4i*)(p);
  v4i hi = *(const v4i*)(p + 16);
  return cat44(lo, hi);
}

// B fragment from packed weights: [frag][lane][8 dwords] -> two b128 loads.
__device__ __forceinline__ v8i load_bfrag(const uint32_t* __restrict__ W,
                                          int frag, int lane) {
  const uint32_t* p = W + ((long)frag * 32 + lane) * 8;
  v4i lo = *(const v4i*)(p);
  v4i hi = *(const v4i*)(p + 4);
  return cat44(lo, hi);
}

// Pack fp32 weight [KD,ND] row-major -> bf16 B-fragment layout (lane-major).
__global__ void k_pack(const float* __restrict__ W, uint32_t* __restrict__ out,
                       int KD, int ND) {
  int t = blockIdx.x * blockDim.x + threadIdx.x;
  int NT = ND >> 4, KT = KD >> 5;
  int lane = t & 31, frag = t >> 5;
  if (frag >= KT * NT) return;
  int kt = frag / NT, nt = frag - kt * NT;
  int n = nt * 16 + (lane & 15);
  int kb = kt * 32 + ((lane >> 4) << 3);
#pragma unroll
  for (int q = 0; q < 8; ++q) {
    int k = kb + 2 * (q & 3) + ((q >> 2) << 4);
    uint32_t lo = f2bf(W[(long)k * ND + n]);
    uint32_t hi = f2bf(W[(long)(k + 1) * ND + n]);
    out[((long)frag * 32 + lane) * 8 + q] = lo | (hi << 16);
  }
}

// out = act(A_bf16[M,KD] @ Wpk + bias). One wave = 32 x ND (2 M-tiles share
// each B fragment).  OBF: emit bf16 output instead of fp32.
template <int KD, int ND, bool RELU, bool OBF>
__global__ __launch_bounds__(128) void k_gemm(const uint16_t* __restrict__ Abf,
                                              const uint32_t* __restrict__ Wpk,
                                              const float* __restrict__ bias,
                                              void* __restrict__ outp, int M) {
  constexpr int KT = KD / 32, NT = ND / 16;
  const int lane = threadIdx.x & 31, wid = threadIdx.x >> 5;
  const int m0 = (blockIdx.x * 4 + wid) * 32;
  if (m0 >= M) return;                       // wave-uniform
  const int row0 = m0 + (lane & 15);
  v8i afr[2][KT];
#pragma unroll
  for (int kt = 0; kt < KT; ++kt) {
    afr[0][kt] = load_afrag(Abf, (long)row0 * KD, kt, lane);
    afr[1][kt] = load_afrag(Abf, (long)(row0 + 16) * KD, kt, lane);
  }
  const int ncol = lane & 15;
  const int mrow = m0 + ((lane >> 4) << 3);
#pragma unroll
  for (int nt = 0; nt < NT; ++nt) {
    float bv = bias[nt * 16 + ncol];
    v8f acc0 = {bv, bv, bv, bv, bv, bv, bv, bv};
    v8f acc1 = acc0;
#pragma unroll
    for (int kt = 0; kt < KT; ++kt) {
      v8i bfr = load_bfrag(Wpk, kt * NT + nt, lane);
      acc0 = wmma_bf16(afr[0][kt], bfr, acc0);
      acc1 = wmma_bf16(afr[1][kt], bfr, acc1);
    }
#pragma unroll
    for (int r = 0; r < 8; ++r) {
      float v0 = acc0[r], v1 = acc1[r];
      if (RELU) { v0 = v0 > 0.f ? v0 : 0.f; v1 = v1 > 0.f ? v1 : 0.f; }
      long o0 = (long)(mrow + r) * ND + nt * 16 + ncol;
      long o1 = o0 + (long)16 * ND;
      if (OBF) {
        ((uint16_t*)outp)[o0] = f2bf(v0);
        ((uint16_t*)outp)[o1] = f2bf(v1);
      } else {
        ((float*)outp)[o0] = v0;
        ((float*)outp)[o1] = v1;
      }
    }
  }
}

// Fused per-node attention; one wave per node; writes o as bf16.
__global__ __launch_bounds__(128) void k_attn(
    const uint16_t* __restrict__ Xb, const float* __restrict__ Q,
    const int* __restrict__ samples, const uint32_t* __restrict__ WkP,
    const float* __restrict__ bk, const uint32_t* __restrict__ WvP,
    const float* __restrict__ bv, uint16_t* __restrict__ Obf, int Nn) {
  __shared__ float kkS[4][16][64];
  __shared__ float vvS[4][16][64];
  __shared__ float scS[4][4][16];
  __shared__ float atS[4][4][KSAMP];
  const int lane = threadIdx.x & 31, w = threadIdx.x >> 5;
  int node = blockIdx.x * 4 + w;
  if (node >= Nn) node = Nn - 1;             // keep EXEC all-ones for WMMA
  const int r15 = lane & 15;
  v8i af[2];
  if (r15 < KSAMP) {
    int idx = samples[(long)node * KSAMP + r15];
#pragma unroll
    for (int kt = 0; kt < 2; ++kt) af[kt] = load_afrag(Xb, (long)idx * 64, kt, lane);
  } else {
    v8i z = {0, 0, 0, 0, 0, 0, 0, 0};
    af[0] = z; af[1] = z;
  }
  const int mbase = (lane >> 4) << 3;
#pragma unroll
  for (int nt = 0; nt < 4; ++nt) {
    float bkv = bk[nt * 16 + r15];
    v8f ak = {bkv, bkv, bkv, bkv, bkv, bkv, bkv, bkv};
    ak = wmma_bf16(af[0], load_bfrag(WkP, 0 * 4 + nt, lane), ak);
    ak = wmma_bf16(af[1], load_bfrag(WkP, 1 * 4 + nt, lane), ak);
    float bvv = bv[nt * 16 + r15];
    v8f av = {bvv, bvv, bvv, bvv, bvv, bvv, bvv, bvv};
    av = wmma_bf16(af[0], load_bfrag(WvP, 0 * 4 + nt, lane), av);
    av = wmma_bf16(af[1], load_bfrag(WvP, 1 * 4 + nt, lane), av);
#pragma unroll
    for (int r = 0; r < 8; ++r) {
      kkS[w][mbase + r][nt * 16 + r15] = ak[r];
      vvS[w][mbase + r][nt * 16 + r15] = av[r];
    }
  }
  __syncthreads();
#pragma unroll
  for (int t = 0; t < 2; ++t) {              // scores q_h . k_j / 4
    int j = lane & 15;
    int h = (lane >> 4) + t * 2;
    if (j < KSAMP) {
      const float* qh = Q + (long)node * 64 + h * 16;
      float s = 0.f;
#pragma unroll
      for (int d = 0; d < 16; ++d) s += qh[d] * kkS[w][j][h * 16 + d];
      scS[w][h][j] = s * 0.25f;
    }
  }
  __syncthreads();
  if (lane < 4) {                            // softmax over 12 keys, head=lane
    int h = lane;
    float mx = -1e30f;
    for (int j = 0; j < KSAMP; ++j) mx = fmaxf(mx, scS[w][h][j]);
    float e[KSAMP], sum = 0.f;
    for (int j = 0; j < KSAMP; ++j) { e[j] = __expf(scS[w][h][j] - mx); sum += e[j]; }
    float inv = 1.f / sum;
    for (int j = 0; j < KSAMP; ++j) atS[w][h][j] = e[j] * inv;
  }
  __syncthreads();
#pragma unroll
  for (int t = 0; t < 2; ++t) {              // o = att . vv  (bf16 out)
    int c = lane + t * 32;
    int h = c >> 4;
    float o = 0.f;
    for (int j = 0; j < KSAMP; ++j) o += atS[w][h][j] * vvS[w][j][c];
    Obf[(long)node * 64 + c] = f2bf(o);
  }
}

// X = LayerNorm(X + Y); also emits bf16 copy of the result (fused cvt).
__global__ __launch_bounds__(256) void k_addln(float* __restrict__ X,
                                               const float* __restrict__ Y,
                                               const float* __restrict__ g,
                                               const float* __restrict__ be,
                                               uint16_t* __restrict__ Xb, int M) {
  int wv = (blockIdx.x * blockDim.x + threadIdx.x) >> 5;
  int lane = threadIdx.x & 31;
  if (wv >= M) return;
  long b = (long)wv * 64;
  float v0 = X[b + lane] + Y[b + lane];
  float v1 = X[b + 32 + lane] + Y[b + 32 + lane];
  float s = v0 + v1;
#pragma unroll
  for (int o = 16; o; o >>= 1) s += __shfl_xor(s, o, 32);
  float mu = s * (1.f / 64.f);
  float d0 = v0 - mu, d1 = v1 - mu;
  float vs = d0 * d0 + d1 * d1;
#pragma unroll
  for (int o = 16; o; o >>= 1) vs += __shfl_xor(vs, o, 32);
  float inv = rsqrtf(vs * (1.f / 64.f) + 1e-5f);
  float o0 = g[lane] * d0 * inv + be[lane];
  float o1 = g[32 + lane] * d1 * inv + be[32 + lane];
  X[b + lane] = o0;       X[b + 32 + lane] = o1;
  Xb[b + lane] = f2bf(o0); Xb[b + 32 + lane] = f2bf(o1);
}

__global__ void k_zero4(float* __restrict__ p, long n4) {
  long t = blockIdx.x * (long)blockDim.x + threadIdx.x;
  if (t >= n4) return;
  float4 z = {0.f, 0.f, 0.f, 0.f};
  *(float4*)(p + t * 4) = z;
}

// x = concat(user,item); A=P=S=x; bf16 copy to Xb. 4 elems/thread.
__global__ void k_init(const float* __restrict__ ue, const float* __restrict__ ie,
                       float* __restrict__ A, float* __restrict__ P,
                       float* __restrict__ S, uint16_t* __restrict__ Xb,
                       int U, long n4) {
  long t = blockIdx.x * (long)blockDim.x + threadIdx.x;
  if (t >= n4) return;
  long i = t * 4;
  long ub = (long)U * 64;
  const float* src = (i < ub) ? (ue + i) : (ie + (i - ub));
  float4 v = *(const float4*)src;
  *(float4*)(A + i) = v; *(float4*)(P + i) = v; *(float4*)(S + i) = v;
  uint2 bb; bb.x = pack2bf(v.x, v.y); bb.y = pack2bf(v.z, v.w);
  *(uint2*)(Xb + i) = bb;
}

// COO spmm: Xn[row] += val * X[col]; thread = (edge, 32-feature half)
__global__ void k_spmm(const float* __restrict__ X, const int* __restrict__ ri,
                       const int* __restrict__ ci, const float* __restrict__ vals,
                       float* __restrict__ Xn, long E2) {
  long t = blockIdx.x * (long)blockDim.x + threadIdx.x;
  if (t >= E2) return;
  long e = t >> 1;
  int half = (int)(t & 1);
  float v = vals[e];
  const float* xs = X + (long)ci[e] * 64 + half * 32;
  float* xd = Xn + (long)ri[e] * 64 + half * 32;
#pragma unroll
  for (int i = 0; i < 32; ++i) atomicAdd(&xd[i], v * xs[i]);
}

// bf16 concat [x | xn] -> [N,128]; 4 cols per thread.
__global__ void k_concat2(const float* __restrict__ X, const float* __restrict__ Xn,
                          uint16_t* __restrict__ Xb, long n32) {
  long t = blockIdx.x * (long)blockDim.x + threadIdx.x;
  if (t >= n32) return;
  long nrow = t >> 5;
  int c = (int)(t & 31) * 4;
  const float* src = (c < 64) ? (X + nrow * 64 + c) : (Xn + nrow * 64 + (c - 64));
  float4 v = *(const float4*)src;
  uint2 bb; bb.x = pack2bf(v.x, v.y); bb.y = pack2bf(v.z, v.w);
  *(uint2*)(Xb + nrow * 128 + c) = bb;
}

// x_new = gnn_out + prev; P = x_new; S += x_new; A = x_new; bf16 copy.
__global__ void k_resid(float* __restrict__ A, const float* __restrict__ B,
                        float* __restrict__ P, float* __restrict__ S,
                        uint16_t* __restrict__ Xb, long n4) {
  long t = blockIdx.x * (long)blockDim.x + threadIdx.x;
  if (t >= n4) return;
  long i = t * 4;
  float4 bb = *(const float4*)(B + i);
  float4 pp = *(const float4*)(P + i);
  float4 a; a.x = bb.x + pp.x; a.y = bb.y + pp.y; a.z = bb.z + pp.z; a.w = bb.w + pp.w;
  *(float4*)(A + i) = a; *(float4*)(P + i) = a;
  float4 ss = *(const float4*)(S + i);
  ss.x += a.x; ss.y += a.y; ss.z += a.z; ss.w += a.w;
  *(float4*)(S + i) = ss;
  uint2 u; u.x = pack2bf(a.x, a.y); u.y = pack2bf(a.z, a.w);
  *(uint2*)(Xb + i) = u;
}

// Build bf16 concat [x | spe_e | de_e | pre_e] -> Xbig [N,256]
__global__ void k_pos(const float* __restrict__ X, const float* __restrict__ deg,
                      const float* __restrict__ pr,
                      const float* __restrict__ sW2, const float* __restrict__ sb1,
                      const float* __restrict__ sb2,
                      const float* __restrict__ dW1, const float* __restrict__ dW2,
                      const float* __restrict__ db1, const float* __restrict__ db2,
                      const float* __restrict__ pW1, const float* __restrict__ pW2,
                      const float* __restrict__ pb1, const float* __restrict__ pb2,
                      uint16_t* __restrict__ Xb, int Nn) {
  long t = blockIdx.x * (long)blockDim.x + threadIdx.x;
  if (t >= (long)Nn * 64) return;
  long n = t >> 6;
  int j = (int)(t & 63);
  float dv = deg[n], pv = pr[n];
  float se = sb2[j], de = db2[j], pe = pb2[j];
  for (int i = 0; i < 32; ++i) {
    float hs = sb1[i];               hs = hs > 0.f ? hs : 0.f;   // spe input = 0
    float hd = dv * dW1[i] + db1[i]; hd = hd > 0.f ? hd : 0.f;
    float hp = pv * pW1[i] + pb1[i]; hp = hp > 0.f ? hp : 0.f;
    se += hs * sW2[i * 64 + j];
    de += hd * dW2[i * 64 + j];
    pe += hp * pW2[i * 64 + j];
  }
  long b = n * 256;
  Xb[b + j]       = f2bf(X[n * 64 + j]);
  Xb[b + 64 + j]  = f2bf(se);
  Xb[b + 128 + j] = f2bf(de);
  Xb[b + 192 + j] = f2bf(pe);
}

// final = (S + comb_out)/4 -> out[0:nd] and out[nd:2nd]  (4 elems/thread)
__global__ void k_final(const float* __restrict__ S, const float* __restrict__ B,
                        float* __restrict__ out, long n) {
  long t = blockIdx.x * (long)blockDim.x + threadIdx.x;
  long i = t * 4;
  if (i >= n) return;
  float4 ss = *(const float4*)(S + i);
  float4 bb = *(const float4*)(B + i);
  float4 f;
  f.x = (ss.x + bb.x) * 0.25f; f.y = (ss.y + bb.y) * 0.25f;
  f.z = (ss.z + bb.z) * 0.25f; f.w = (ss.w + bb.w) * 0.25f;
  *(float4*)(out + i) = f;
  *(float4*)(out + n + i) = f;
}

// ---------------------------------------------------------------------------
static inline int cdiv(long a, long b) { return (int)((a + b - 1) / b); }

extern "C" void kernel_launch(void* const* d_in, const int* in_sizes, int n_in,
                              void* d_out, int out_size, void* d_ws, size_t ws_size,
                              hipStream_t stream) {
  // jax pytree order (sorted dict keys, list order):
  // 0-3: gnn[0].{W,b}, gnn[1].{W,b}; 4: item_emb;
  // 5-8: comb{W1,W2,b1,b2}; 9-12: de; 13-16: pre; 17-20: spe;
  // 21+16l: tf[l] = {Wf1,Wf2,Wk,Wo,Wq,Wv,be1,be2,bf1,bf2,bk,bo,bq,bv,g1,g2};
  // 69: user_emb; 70: adj_vals; 71: degrees; 72: pagerank;
  // 73: attention_samples; 74: adj_row; 75: adj_col
  auto F   = [&](int i) { return (const float*)d_in[i]; };
  auto I32 = [&](int i) { return (const int*)d_in[i]; };
  (void)n_in; (void)out_size; (void)ws_size;

  const int  U  = in_sizes[69] / 64;
  const int  N  = U + in_sizes[4] / 64;
  const long E  = in_sizes[70];
  const long nd = (long)N * 64;

  // workspace carve-up (256B aligned so all b128 accesses are 16B aligned)
  char* cur = (char*)d_ws;
  auto carve = [&](size_t bytes) {
    char* p = cur; cur += (bytes + 255) & ~(size_t)255; return p;
  };
  uint32_t* PK   = (uint32_t*)carve(110000 * 4);        // packed weights
  float*    A    = (float*)carve(nd * 4);               // current x
  float*    P    = (float*)carve(nd * 4);               // last elist entry
  float*    S    = (float*)carve(nd * 4);               // elist running sum
  float*    B    = (float*)carve(nd * 4);               // scratch y
  float*    C    = (float*)carve((size_t)N * 256 * 4);  // Q / y / xn scratch
  uint16_t* Xb1  = (uint16_t*)carve((size_t)N * 128 * 2); // x / concat bf16
  uint16_t* Xbig = (uint16_t*)carve((size_t)N * 256 * 2); // hidden / pos / o bf16

  uint32_t* pk_cur = PK;
  auto palloc = [&](int kd, int ndm) { uint32_t* p = pk_cur; pk_cur += (long)kd * ndm / 2; return p; };
  auto pack = [&](const float* W, uint32_t* dst, int kd, int ndm) {
    int th = (kd / 32) * (ndm / 16) * 32;
    k_pack<<<cdiv(th, 128), 128, 0, stream>>>(W, dst, kd, ndm);
  };

  uint32_t *tq[3], *tk[3], *tv[3], *to_[3], *tf1[3], *tf2[3], *gw[2], *cw1, *cw2;
  for (int l = 0; l < 3; ++l) {
    int ib = 21 + 16 * l;
    tq[l]  = palloc(64, 64);   pack(F(ib + 4), tq[l],  64, 64);
    tk[l]  = palloc(64, 64);   pack(F(ib + 2), tk[l],  64, 64);
    tv[l]  = palloc(64, 64);   pack(F(ib + 5), tv[l],  64, 64);
    to_[l] = palloc(64, 64);   pack(F(ib + 3), to_[l], 64, 64);
    tf1[l] = palloc(64, 256);  pack(F(ib + 0), tf1[l], 64, 256);
    tf2[l] = palloc(256, 64);  pack(F(ib + 1), tf2[l], 256, 64);
  }
  for (int l = 0; l < 2; ++l) { gw[l] = palloc(128, 64); pack(F(2 * l), gw[l], 128, 64); }
  cw1 = palloc(256, 128); pack(F(5), cw1, 256, 128);
  cw2 = palloc(128, 64);  pack(F(6), cw2, 128, 64);

  const int gemmBlocks = cdiv(cdiv(N, 32), 4);          // 32-row waves, 4/block
  const int lnBlocks   = cdiv(N, 8);                    // 8 waves / block

  k_init<<<cdiv(nd / 4, 256), 256, 0, stream>>>(F(69), F(4), A, P, S, Xb1, U, nd / 4);

  auto run_tf = [&](int l) {
    int ib = 21 + 16 * l;
    // q = x@Wq+bq -> C
    k_gemm<64, 64, false, false><<<gemmBlocks, 128, 0, stream>>>(Xb1, tq[l], F(ib + 12), C, N);
    // fused attention -> o bf16 in Xbig[:nd]
    k_attn<<<cdiv(N, 4), 128, 0, stream>>>(Xb1, C, I32(73), tk[l], F(ib + 10),
                                           tv[l], F(ib + 13), Xbig, N);
    // y = o@Wo+bo -> C ; x = LN(x+y) -> A (+bf16 Xb1)
    k_gemm<64, 64, false, false><<<gemmBlocks, 128, 0, stream>>>(Xbig, to_[l], F(ib + 11), C, N);
    k_addln<<<lnBlocks, 256, 0, stream>>>(A, C, F(ib + 14), F(ib + 6), Xb1, N);
    // h = relu(x@Wf1+bf1) -> bf16 Xbig ; y = h@Wf2+bf2 -> B ; LN -> A (+Xb1)
    k_gemm<64, 256, true, true><<<gemmBlocks, 128, 0, stream>>>(Xb1, tf1[l], F(ib + 8), Xbig, N);
    k_gemm<256, 64, false, false><<<gemmBlocks, 128, 0, stream>>>(Xbig, tf2[l], F(ib + 9), B, N);
    k_addln<<<lnBlocks, 256, 0, stream>>>(A, B, F(ib + 15), F(ib + 7), Xb1, N);
  };

  for (int l = 0; l < 2; ++l) {
    run_tf(l);
    k_zero4<<<cdiv(nd / 4, 256), 256, 0, stream>>>(C, nd / 4);
    k_spmm<<<cdiv(E * 2, 256), 256, 0, stream>>>(A, I32(74), I32(75), F(70), C, E * 2);
    k_concat2<<<cdiv((long)N * 32, 256), 256, 0, stream>>>(A, C, Xb1, (long)N * 32);
    k_gemm<128, 64, true, false><<<gemmBlocks, 128, 0, stream>>>(Xb1, gw[l], F(2 * l + 1), B, N);
    k_resid<<<cdiv(nd / 4, 256), 256, 0, stream>>>(A, B, P, S, Xb1, nd / 4);
  }
  run_tf(2);

  k_pos<<<cdiv(nd, 256), 256, 0, stream>>>(A, F(71), F(72),
                                           F(18), F(19), F(20),          // spe W2,b1,b2
                                           F(9), F(10), F(11), F(12),    // de
                                           F(13), F(14), F(15), F(16),   // pre
                                           Xbig, N);
  k_gemm<256, 128, true, true><<<gemmBlocks, 128, 0, stream>>>(Xbig, cw1, F(7), Xb1, N);
  k_gemm<128, 64, false, false><<<gemmBlocks, 128, 0, stream>>>(Xb1, cw2, F(8), B, N);
  k_final<<<cdiv(nd / 4, 256), 256, 0, stream>>>(S, B, (float*)d_out, nd);
}
